// BidirectionalLSTMModel_64793876627694
// MI455X (gfx1250) — matrix operand order repaired and checked by reading, not compiled, not run on hardware
//
#include <hip/hip_runtime.h>
#include <cstdint>

// ---------------------------------------------------------------------------
// Bidirectional LSTM for MI455X (gfx1250, wave32, WMMA).
//   V=32000, E=512, H=1024, L=2048.
// Strategy:
//   1) repack [Wi|Wf|Wg|Wo] -> bf16 Wcat[2][1536][4096], biases -> bcat[2][4096]
//   2) gather embeddings -> bf16 emb[2048][512]
//   3) Zx[dir][t][4096] = emb(dir,t) @ Wcat[rows 0..512) + bcat   (WMMA GEMM)
//   4) persistent recurrence: 2 blocks (one per direction), h kept in LDS,
//      zh = h @ Wcat[rows 512..1536) via WMMA with h broadcast across A rows
//   5) out = hs(bf16) @ Wout + bout: 128x128 block tiles, 8 WMMA per K-stage,
//      A tile staged with GLOBAL_LOAD_ASYNC_TO_LDS (ASYNCcnt), B tile converted
//      fp32->bf16 while staged transposed, next tile prefetched.
// ---------------------------------------------------------------------------

#define L_SEQ 2048
#define E_DIM 512
#define H_DIM 1024
#define V_DIM 32000
#define EH    1536   // E + H
#define G4    4096   // 4*H
#define H2    2048   // 2*H

typedef __attribute__((ext_vector_type(16))) __bf16 v16bf;
typedef __attribute__((ext_vector_type(8)))  __bf16 v8bf;
typedef __attribute__((ext_vector_type(8)))  float  v8f;

static __device__ __forceinline__ v16bf cat8(v8bf lo, v8bf hi) {
  return __builtin_shufflevector(lo, hi, 0,1,2,3,4,5,6,7,8,9,10,11,12,13,14,15);
}
static __device__ __forceinline__ float sigmoidf_(float x) {
  return 1.0f / (1.0f + __expf(-x));
}
// CDNA5 async copy: memory -> LDS, 8 bytes per lane, tracked with ASYNCcnt.
static __device__ __forceinline__ void async_ld_lds_b64(uint32_t lds_addr,
                                                        uint64_t gaddr) {
  asm volatile("global_load_async_to_lds_b64 %0, %1, off"
               :: "v"(lds_addr), "v"(gaddr) : "memory");
}
static __device__ __forceinline__ void wait_asynccnt0() {
  asm volatile("s_wait_asynccnt 0x0" ::: "memory");
}

// ---------------- kernel 1: repack weights + biases to bf16 ----------------
__global__ __launch_bounds__(256)
void prep_weights_kernel(const float* Wi_f, const float* bi_f,
                         const float* Wf_f, const float* bf_f,
                         const float* Wg_f, const float* bg_f,
                         const float* Wo_f, const float* bo_f,
                         const float* Wi_b, const float* bi_b,
                         const float* Wf_b, const float* bf_b,
                         const float* Wg_b, const float* bg_b,
                         const float* Wo_b, const float* bo_b,
                         __bf16* Wcat, float* bcat) {
  const int gid = blockIdx.x * 256 + threadIdx.x;
  const int total = 2 * EH * G4;
  if (gid < total) {
    const int dir = gid / (EH * G4);
    const int rem = gid - dir * (EH * G4);
    const int r    = rem >> 12;          // row in [0,1536)
    const int c4   = rem & (G4 - 1);     // gate-concat column
    const int gate = c4 >> 10;
    const int col  = c4 & (H_DIM - 1);
    const int sel = dir * 4 + gate;
    const float* W =
        sel == 0 ? Wi_f : sel == 1 ? Wf_f : sel == 2 ? Wg_f : sel == 3 ? Wo_f :
        sel == 4 ? Wi_b : sel == 5 ? Wf_b : sel == 6 ? Wg_b : Wo_b;
    Wcat[gid] = (__bf16)W[r * H_DIM + col];
  }
  if (gid < 2 * G4) {
    const int dir  = gid >> 12;
    const int c4   = gid & (G4 - 1);
    const int gate = c4 >> 10;
    const int col  = c4 & (H_DIM - 1);
    const int sel = dir * 4 + gate;
    const float* b =
        sel == 0 ? bi_f : sel == 1 ? bf_f : sel == 2 ? bg_f : sel == 3 ? bo_f :
        sel == 4 ? bi_b : sel == 5 ? bf_b : sel == 6 ? bg_b : bo_b;
    bcat[gid] = b[col];
  }
}

// ---------------- kernel 2: embedding gather -> bf16 -----------------------
__global__ __launch_bounds__(256)
void embed_gather_kernel(const long long* __restrict__ seq,
                         const float* __restrict__ embW,
                         __bf16* __restrict__ emb) {
  const int gid = blockIdx.x * 256 + threadIdx.x;   // L*E total, exact grid
  const int t = gid >> 9;
  const int e = gid & (E_DIM - 1);
  const long long row = seq[t];
  emb[gid] = (__bf16)embW[(size_t)row * E_DIM + e];
}

// ---------------- kernel 3: Zx = emb @ Wx + b (WMMA GEMM) ------------------
// grid: (G4/128, L/16, 2 dirs), block 256 (8 waves, one 16x16 N-tile each).
__global__ __launch_bounds__(256)
void zx_gemm_kernel(const __bf16* __restrict__ emb,
                    const __bf16* __restrict__ Wcat,
                    const float* __restrict__ bcat,
                    float* __restrict__ Zx) {
  __shared__ __align__(16) __bf16 ldsA[16][E_DIM];  // 16 KB, whole K once
  __shared__ __align__(16) __bf16 ldsBT[128][32];   // 8 KB, transposed B tile
  const int tid  = threadIdx.x;
  const int lane = tid & 31;
  const int w    = tid >> 5;
  const int dir   = blockIdx.z;
  const int tBase = blockIdx.y * 16;
  const int nBase = blockIdx.x * 128;
  const __bf16* Wd = Wcat + (size_t)dir * EH * G4;

  for (int ii = 0; ii < 32; ++ii) {                 // stage A 16x512
    const int idx = ii * 256 + tid;
    const int r = idx >> 9, k = idx & (E_DIM - 1);
    const int srcRow = (dir == 0) ? (tBase + r) : (L_SEQ - 1 - (tBase + r));
    ldsA[r][k] = emb[(size_t)srcRow * E_DIM + k];
  }

  const int m   = lane & 15;
  const int kb  = (lane < 16) ? 0 : 16;   // A-frag K layout (wave32)
  const int kb2 = (lane < 16) ? 0 : 16;   // B-frag K layout
  const int nl  = w * 16 + (lane & 15);

  v8f acc = {};
  for (int ks = 0; ks < E_DIM / 32; ++ks) {
    const int kBase = ks * 32;
    __syncthreads();
    for (int ii = 0; ii < 16; ++ii) {               // stage B^T 32x128
      const int idx = ii * 256 + tid;
      const int kk = idx >> 7, n = idx & 127;
      ldsBT[n][kk] = Wd[(size_t)(kBase + kk) * G4 + nBase + n];
    }
    __syncthreads();
    const v16bf a = cat8(*(const v8bf*)&ldsA[m][kBase + kb],
                         *(const v8bf*)&ldsA[m][kBase + kb + 8]);
    const v16bf b = cat8(*(const v8bf*)&ldsBT[nl][kb2],
                         *(const v8bf*)&ldsBT[nl][kb2 + 8]);
    acc = __builtin_amdgcn_wmma_f32_16x16x32_bf16(false, a, false, b,
                                                  (short)0, acc, false, false);
  }

  const int N = nBase + w * 16 + (lane & 15);
  const float bias = bcat[dir * G4 + N];
  float* Zd = Zx + (size_t)dir * L_SEQ * G4;
#pragma unroll
  for (int r = 0; r < 8; ++r) {
    const int M = r + ((lane < 16) ? 0 : 8);        // C/D layout
    Zd[(size_t)(tBase + M) * G4 + N] = acc[r] + bias;
  }
}

// ---------------- kernel 4: persistent recurrence --------------------------
// 2 blocks (dir), 1024 threads = 32 waves. h lives in LDS; zh matvec via WMMA
// with h broadcast across A rows (all D rows identical -> take acc[.][0]).
__global__ __launch_bounds__(1024)
void lstm_rec_kernel(const __bf16* __restrict__ Wcat,
                     const float* __restrict__ Zx,
                     __bf16* __restrict__ hs) {
  __shared__ __align__(16) __bf16 h_lds[H_DIM];     // 2 KB
  __shared__ float zh_lds[G4];                      // 16 KB
  const int tid  = threadIdx.x;
  const int lane = tid & 31;
  const int w    = tid >> 5;                        // 0..31
  const int dir  = blockIdx.x;
  const __bf16* Wd = Wcat + (size_t)dir * EH * G4 + (size_t)E_DIM * G4; // Wh rows
  const float*  Zd = Zx + (size_t)dir * L_SEQ * G4;

  float c = 0.0f;
  h_lds[tid & (H_DIM - 1)] = (__bf16)0.0f;

  const int kb     = (lane < 16) ? 0 : 16;
  const int kb2off = (lane < 16) ? 0 : 16;
  const int nlane  = lane & 15;

  for (int t = 0; t < L_SEQ; ++t) {
    __syncthreads();                                // h_lds ready / zh_lds free
    v8f acc[8] = {};
    for (int ks = 0; ks < H_DIM / 32; ++ks) {
      const int kBase = ks * 32;
      const v16bf a = cat8(*(const v8bf*)&h_lds[kBase + kb],
                           *(const v8bf*)&h_lds[kBase + kb + 8]);
      const int krow = kBase + kb2off;
#pragma unroll
      for (int j = 0; j < 8; ++j) {
        const int n = w * 128 + j * 16 + nlane;
        const __bf16* bp = Wd + (size_t)krow * G4 + n;
        v16bf b;
#pragma unroll
        for (int jj = 0; jj < 16; ++jj) b[jj] = bp[(size_t)jj * G4];
        acc[j] = __builtin_amdgcn_wmma_f32_16x16x32_bf16(false, a, false, b,
                                                         (short)0, acc[j],
                                                         false, false);
      }
    }
    if (lane < 16) {
#pragma unroll
      for (int j = 0; j < 8; ++j)
        zh_lds[w * 128 + j * 16 + lane] = acc[j][0];
    }
    __syncthreads();                                // zh ready, h reads done
    const size_t zb = (size_t)t * G4;
    const float zi = Zd[zb + tid]             + zh_lds[tid];
    const float zf = Zd[zb + H_DIM + tid]     + zh_lds[H_DIM + tid];
    const float zg = Zd[zb + 2 * H_DIM + tid] + zh_lds[2 * H_DIM + tid];
    const float zo = Zd[zb + 3 * H_DIM + tid] + zh_lds[3 * H_DIM + tid];
    c = sigmoidf_(zf) * c + sigmoidf_(zi) * tanhf(zg);
    const float h = sigmoidf_(zo) * tanhf(c);
    h_lds[tid] = (__bf16)h;
    const int row = (dir == 0) ? t : (L_SEQ - 1 - t);
    hs[(size_t)row * H2 + dir * H_DIM + tid] = (__bf16)h;
  }
}

// ---------------- kernel 5: out = hs @ Wout + bout (WMMA GEMM) -------------
// grid: (V/128, L/128), block 256 (8 waves). Each wave owns a 16-column strip
// and 8 M-tiles (128 rows) -> 8 WMMAs per K-stage reusing one B fragment.
// A tile (bf16 in memory) staged via GLOBAL_LOAD_ASYNC_TO_LDS_B64 (ASYNCcnt);
// B tile converted fp32->bf16 while staged transposed.
__global__ __launch_bounds__(256)
void out_gemm_kernel(const __bf16* __restrict__ hs,
                     const float* __restrict__ Wout,
                     const float* __restrict__ bout,
                     float* __restrict__ out) {
  __shared__ __align__(16) __bf16 ldsA[128][32];    // 8 KB
  __shared__ __align__(16) __bf16 ldsBT[128][32];   // 8 KB
  const int tid = threadIdx.x, lane = tid & 31, w = tid >> 5;
  const int nBase = blockIdx.x * 128;
  const int mBase = blockIdx.y * 128;
  const int m16 = lane & 15;
  const int kb  = (lane < 16) ? 0 : 16;
  const int kb2 = (lane < 16) ? 0 : 16;
  const int nl  = w * 16 + (lane & 15);
  const char* hsb = (const char*)hs;                 // byte view, row = 4096 B
  const uint32_t ldsA_base = (uint32_t)(uintptr_t)&ldsA[0][0];

  v8f acc[8] = {};
  for (int kt = 0; kt < H2 / 32; ++kt) {
    const int kBase = kt * 32;
    __syncthreads();                                 // prior fragment reads done
    // stage A 128x32 bf16 (8 KB) via async copy: 4 x (256 lanes x 8 B)
#pragma unroll
    for (int i = 0; i < 4; ++i) {
      const int o = i * 2048 + tid * 8;              // flat byte offset in tile
      const int row = o >> 6, colb = o & 63;         // 64 B per tile row
      async_ld_lds_b64(ldsA_base + (uint32_t)o,
                       (uint64_t)(uintptr_t)(hsb + (size_t)(mBase + row) * (H2 * 2)
                                                 + (size_t)kBase * 2 + colb));
    }
    // stage B^T 32x128 with fp32->bf16 conversion
#pragma unroll
    for (int ii = 0; ii < 16; ++ii) {
      const int idx = ii * 256 + tid;
      const int kk = idx >> 7, n = idx & 127;
      ldsBT[n][kk] = (__bf16)Wout[(size_t)(kBase + kk) * V_DIM + nBase + n];
      if (ii == 0 && kt + 1 < H2 / 32)               // prefetch next B tile
        __builtin_prefetch(&Wout[(size_t)(kBase + 32 + kk) * V_DIM + nBase + n], 0, 1);
    }
    wait_asynccnt0();                                // A tile landed in LDS
    __syncthreads();
    const v16bf b = cat8(*(const v8bf*)&ldsBT[nl][kb2],
                         *(const v8bf*)&ldsBT[nl][kb2 + 8]);
#pragma unroll
    for (int r = 0; r < 8; ++r) {                    // 8 M-tiles share B frag
      const v16bf a = cat8(*(const v8bf*)&ldsA[r * 16 + m16][kb],
                           *(const v8bf*)&ldsA[r * 16 + m16][kb + 8]);
      acc[r] = __builtin_amdgcn_wmma_f32_16x16x32_bf16(false, a, false, b,
                                                       (short)0, acc[r],
                                                       false, false);
    }
  }
  const int N = nBase + w * 16 + (lane & 15);
  const float bb = bout[N];
  const int hi8 = (lane < 16) ? 0 : 8;
#pragma unroll
  for (int r = 0; r < 8; ++r)
#pragma unroll
    for (int j = 0; j < 8; ++j)
      out[(size_t)(mBase + r * 16 + j + hi8) * V_DIM + N] = acc[r][j] + bb;
}

// ---------------------------------------------------------------------------
extern "C" void kernel_launch(void* const* d_in, const int* in_sizes, int n_in,
                              void* d_out, int out_size, void* d_ws, size_t ws_size,
                              hipStream_t stream) {
  (void)in_sizes; (void)n_in; (void)out_size; (void)ws_size;
  const long long* seq  = (const long long*)d_in[0];
  const float* embW = (const float*)d_in[1];
  const float* Wi_f = (const float*)d_in[2];  const float* bi_f = (const float*)d_in[3];
  const float* Wf_f = (const float*)d_in[4];  const float* bf_f = (const float*)d_in[5];
  const float* Wg_f = (const float*)d_in[6];  const float* bg_f = (const float*)d_in[7];
  const float* Wo_f = (const float*)d_in[8];  const float* bo_f = (const float*)d_in[9];
  const float* Wi_b = (const float*)d_in[10]; const float* bi_b = (const float*)d_in[11];
  const float* Wf_b = (const float*)d_in[12]; const float* bf_b = (const float*)d_in[13];
  const float* Wg_b = (const float*)d_in[14]; const float* bg_b = (const float*)d_in[15];
  const float* Wo_b = (const float*)d_in[16]; const float* bo_b = (const float*)d_in[17];
  const float* Wout = (const float*)d_in[18]; const float* bout = (const float*)d_in[19];
  float* out = (float*)d_out;

  // workspace layout (all chunk sizes are multiples of 256 B); total ~103 MB
  char* ws = (char*)d_ws;
  size_t off = 0;
  __bf16* emb_bf = (__bf16*)(ws + off); off += (size_t)L_SEQ * E_DIM * 2;      // 2 MB
  __bf16* Wcat   = (__bf16*)(ws + off); off += (size_t)2 * EH * G4 * 2;        // 25 MB
  float*  bcat   = (float*)(ws + off);  off += (size_t)2 * G4 * 4;             // 32 KB
  float*  Zx     = (float*)(ws + off);  off += (size_t)2 * L_SEQ * G4 * 4;     // 67 MB
  __bf16* hsbuf  = (__bf16*)(ws + off);                                        // 8.4 MB

  prep_weights_kernel<<<(2 * EH * G4) / 256, 256, 0, stream>>>(
      Wi_f, bi_f, Wf_f, bf_f, Wg_f, bg_f, Wo_f, bo_f,
      Wi_b, bi_b, Wf_b, bf_b, Wg_b, bg_b, Wo_b, bo_b, Wcat, bcat);
  embed_gather_kernel<<<(L_SEQ * E_DIM) / 256, 256, 0, stream>>>(seq, embW, emb_bf);
  zx_gemm_kernel<<<dim3(G4 / 128, L_SEQ / 16, 2), 256, 0, stream>>>(emb_bf, Wcat, bcat, Zx);
  lstm_rec_kernel<<<2, 1024, 0, stream>>>(Wcat, Zx, hsbuf);
  out_gemm_kernel<<<dim3(V_DIM / 128, L_SEQ / 128), 256, 0, stream>>>(hsbuf, Wout, bout, out);
}
